// WorkspaceController_47708496724659
// MI455X (gfx1250) — matrix operand order repaired and checked by reading, not compiled, not run on hardware
//
#include <hip/hip_runtime.h>
#include <hip/hip_bf16.h>
#include <math.h>

// Sizes from the reference
#define B_    4096
#define S_    32
#define D_    512
#define V_    32000
#define H_    256

typedef __attribute__((ext_vector_type(16))) _Float16 v16h;
typedef __attribute__((ext_vector_type(8)))  _Float16 v8h;
typedef __attribute__((ext_vector_type(8)))  float    v8f;

// ---------------------------------------------------------------- zero out[6]
__global__ void wc_zero6(float* out) {
    if (threadIdx.x < 6) out[threadIdx.x] = 0.0f;
}

// ------------------------------------------------- mean softmax entropy -> out[4]
// One block (256 threads = 8 waves) per batch row. Online (m, Z, T) merge:
//   Z = sum exp(l-m), T = sum l*exp(l-m), entropy = log(Z) - T/Z + m
// float4 loads (global_load_b128); one max-rescale check per 4 elements.
__global__ __launch_bounds__(256)
void wc_entropy(const float* __restrict__ logits, float* __restrict__ out) {
    const int b = blockIdx.x;
    const float4* row4 = (const float4*)(logits + (size_t)b * V_);

    float m = -INFINITY, Z = 0.0f, T = 0.0f;
    for (int i = threadIdx.x; i < V_ / 4; i += 256) {
        float4 v = row4[i];
        float m4 = fmaxf(fmaxf(v.x, v.y), fmaxf(v.z, v.w));
        if (m4 > m) {                 // rare after warm-up
            float s = __expf(m - m4); // exp(-inf)=0 handles first iteration
            Z *= s; T *= s; m = m4;
        }
        float e0 = __expf(v.x - m);
        float e1 = __expf(v.y - m);
        float e2 = __expf(v.z - m);
        float e3 = __expf(v.w - m);
        Z += (e0 + e1) + (e2 + e3);
        T += (v.x * e0 + v.y * e1) + (v.z * e2 + v.w * e3);
    }
    // wave32 shuffle merge
    #pragma unroll
    for (int msk = 16; msk >= 1; msk >>= 1) {
        float m2 = __shfl_xor(m, msk, 32);
        float Z2 = __shfl_xor(Z, msk, 32);
        float T2 = __shfl_xor(T, msk, 32);
        float mm = fmaxf(m, m2);
        float s1 = __expf(m - mm), s2 = __expf(m2 - mm);
        Z = Z * s1 + Z2 * s2;
        T = T * s1 + T2 * s2;
        m = mm;
    }
    __shared__ float sm[8], sZ[8], sT[8];
    const int wv = threadIdx.x >> 5, ln = threadIdx.x & 31;
    if (ln == 0) { sm[wv] = m; sZ[wv] = Z; sT[wv] = T; }
    __syncthreads();
    if (threadIdx.x == 0) {
        m = sm[0]; Z = sZ[0]; T = sT[0];
        #pragma unroll
        for (int i = 1; i < 8; i++) {
            float mm = fmaxf(m, sm[i]);
            float s1 = __expf(m - mm), s2 = __expf(sm[i] - mm);
            Z = Z * s1 + sZ[i] * s2;
            T = T * s1 + sT[i] * s2;
            m = mm;
        }
        float ent = __logf(Z) - T / Z + m;
        atomicAdd(out + 4, ent * (1.0f / (float)B_));
    }
}

// -------------------------- slot norms -> out[5], x = mean_S(slots) -> ws
// One block (8 waves) per batch index b. Wave w owns s-rows w*4..w*4+3.
// Lane L covers columns d = L*16 .. L*16+15 (contiguous 64B per lane).
__global__ __launch_bounds__(256)
void wc_slots(const float* __restrict__ slots, float* __restrict__ xout,
              float* __restrict__ out) {
    const int b  = blockIdx.x;
    const int w  = threadIdx.x >> 5;
    const int ln = threadIdx.x & 31;

    __shared__ float xsh[8][D_];   // per-wave column partial sums (16 KB)

    float colsum[16];
    #pragma unroll
    for (int k = 0; k < 16; k++) colsum[k] = 0.0f;
    float normpart = 0.0f;

    #pragma unroll
    for (int i = 0; i < 4; i++) {
        const int s = w * 4 + i;
        const float* row = slots + ((size_t)b * S_ + s) * D_ + ln * 16;
        float sq = 0.0f;
        #pragma unroll
        for (int k = 0; k < 16; k++) {
            float v = row[k];
            colsum[k] += v;
            sq += v * v;
        }
        #pragma unroll
        for (int msk = 16; msk >= 1; msk >>= 1) sq += __shfl_xor(sq, msk, 32);
        if (ln == 0) normpart += sqrtf(sq);
    }
    if (ln == 0)
        atomicAdd(out + 5, normpart * (1.0f / ((float)B_ * (float)S_)));

    #pragma unroll
    for (int k = 0; k < 16; k++) xsh[w][ln * 16 + k] = colsum[k];
    __syncthreads();

    for (int d = threadIdx.x; d < D_; d += 256) {
        float sum = 0.0f;
        #pragma unroll
        for (int ww = 0; ww < 8; ww++) sum += xsh[ww][d];
        xout[(size_t)b * D_ + d] = sum * (1.0f / (float)S_);
    }
}

// ----------------- LN -> WMMA GEMM (xn @ w1^T) -> GELU -> @ w2^T -> sigmoid -> mean
// 256 blocks, each handles 16 batch rows. 8 waves; wave w computes N-tiles
// n0 = 16*w and 16*(w+8) of the [16 x 256] hidden tile via v_wmma_f32_16x16x32_f16.
__global__ __launch_bounds__(256)
void wc_mlp(const float* __restrict__ x,
            const float* __restrict__ ln_w, const float* __restrict__ ln_b,
            const float* __restrict__ w1,   const float* __restrict__ b1,
            const float* __restrict__ w2,   const float* __restrict__ b2,
            float* __restrict__ out) {
    const int blk = blockIdx.x;
    const int tid = threadIdx.x;
    const int w   = tid >> 5;
    const int ln  = tid & 31;
    const int nl  = ln & 15;        // local N / local M
    const int hi  = ln >> 4;        // lane half

    __shared__ __align__(16) _Float16 Ash[16][D_];   // LN'd f16 A tile (16 KB)
    __shared__ float w2sh[4][H_];                    // w2 staged (4 KB)
    __shared__ float hpart[16][4];                   // second-GEMM partials

    // stage w2
    #pragma unroll
    for (int j = 0; j < 4; j++) w2sh[j][tid] = w2[j * H_ + tid];
    if (tid < 64) hpart[tid >> 2][tid & 3] = 0.0f;

    // LayerNorm: wave w handles block-local rows 2w, 2w+1; lane covers 16 cols
    #pragma unroll
    for (int i = 0; i < 2; i++) {
        const int m  = w * 2 + i;
        const int gm = blk * 16 + m;
        const float* xr = x + (size_t)gm * D_ + ln * 16;
        float vals[16], sum = 0.0f, sumsq = 0.0f;
        #pragma unroll
        for (int q = 0; q < 16; q++) {
            float v = xr[q];
            vals[q] = v; sum += v; sumsq += v * v;
        }
        #pragma unroll
        for (int msk = 16; msk >= 1; msk >>= 1) {
            sum   += __shfl_xor(sum,   msk, 32);
            sumsq += __shfl_xor(sumsq, msk, 32);
        }
        const float mu   = sum * (1.0f / (float)D_);
        const float var  = sumsq * (1.0f / (float)D_) - mu * mu;
        const float rstd = rsqrtf(var + 1e-5f);
        #pragma unroll
        for (int q = 0; q < 16; q++) {
            const int d = ln * 16 + q;
            float xn = (vals[q] - mu) * rstd * ln_w[d] + ln_b[d];
            Ash[m][d] = (_Float16)xn;
        }
    }
    __syncthreads();

    // WMMA GEMM + GELU + 256->4 projection
    const int kbase = hi * 8;    // A-operand K base for this lane half
    const int koffb = hi * 8;    // B-operand K base for this lane half
    const int ma    = nl;        // A/C local row-col index

    for (int t = w; t < 16; t += 8) {
        const int n0 = t * 16;
        const int n  = n0 + nl;
        v8f acc = {};
        for (int kc = 0; kc < 16; kc++) {
            // A fragment from LDS: halves [kbase..kbase+7] ++ [kbase+16..kbase+23]
            const _Float16* ap = &Ash[ma][kc * 32 + kbase];
            v8h alo = *(const v8h*)ap;
            v8h ahi = *(const v8h*)(ap + 16);
            v16h a = __builtin_shufflevector(alo, ahi,
                     0,1,2,3,4,5,6,7, 8,9,10,11,12,13,14,15);
            // B fragment: column n of w1^T == row n of w1, K contiguous
            const float* wp = w1 + (size_t)n * D_ + kc * 32 + koffb;
            v16h bf;
            #pragma unroll
            for (int e = 0; e < 16; e++) bf[e] = (_Float16)wp[(e & 7) + ((e >> 3) << 4)];
            acc = __builtin_amdgcn_wmma_f32_16x16x32_f16(
                      false, a, false, bf, (short)0, acc, false, false);
        }
        const float b1v = b1[n];
        #pragma unroll
        for (int r = 0; r < 8; r++) {
            const int m = r + 8 * hi;          // C layout: VGPR r, lane half
            float v = acc[r] + b1v;
            float g = 0.5f * v * (1.0f + erff(v * 0.70710678118654752f)); // exact GELU
            #pragma unroll
            for (int j = 0; j < 4; j++) {
                float p = g * w2sh[j][n];
                #pragma unroll
                for (int msk = 8; msk >= 1; msk >>= 1)
                    p += __shfl_xor(p, msk, 16);  // reduce over 16-lane group
                if (nl == 0) atomicAdd(&hpart[m][j], p);
            }
        }
    }
    __syncthreads();

    if (tid < 64) {
        const int m = tid >> 2, j = tid & 3;
        float val = hpart[m][j] + b2[j];
        float p = 1.0f / (1.0f + __expf(-val));
        atomicAdd(out + j, p * (1.0f / (float)B_));
    }
}

// ---------------------------------------------------------------- launcher
extern "C" void kernel_launch(void* const* d_in, const int* in_sizes, int n_in,
                              void* d_out, int out_size, void* d_ws, size_t ws_size,
                              hipStream_t stream) {
    const float* slots  = (const float*)d_in[0];
    const float* logits = (const float*)d_in[1];
    const float* ln_w   = (const float*)d_in[2];
    const float* ln_b   = (const float*)d_in[3];
    const float* w1     = (const float*)d_in[4];
    const float* b1     = (const float*)d_in[5];
    const float* w2     = (const float*)d_in[6];
    const float* b2     = (const float*)d_in[7];
    float* out = (float*)d_out;
    float* xbuf = (float*)d_ws;   // [B_, D_] = 8 MB

    wc_zero6<<<1, 32, 0, stream>>>(out);
    wc_entropy<<<B_, 256, 0, stream>>>(logits, out);
    wc_slots<<<B_, 256, 0, stream>>>(slots, xbuf, out);
    wc_mlp<<<B_ / 16, 256, 0, stream>>>(xbuf, ln_w, ln_b, w1, b1, w2, b2, out);
}
